// GateDotProductAttention_50070728737612
// MI455X (gfx1250) — matrix operand order repaired and checked by reading, not compiled, not run on hardware
//
#include <hip/hip_runtime.h>
#include <math.h>

// CDNA5 wave32 WMMA types
typedef __attribute__((ext_vector_type(16))) __bf16 v16bf;
typedef __attribute__((ext_vector_type(8)))  float  v8f;

#define WMMA_BF16(a, b, c) \
  __builtin_amdgcn_wmma_f32_16x16x32_bf16(false, (a), false, (b), (short)0, (c), false, false)

#if defined(__has_builtin)
#if __has_builtin(__builtin_amdgcn_s_wait_asynccnt)
#define WAIT_ASYNC(N) __builtin_amdgcn_s_wait_asynccnt(N)
#endif
#endif
#ifndef WAIT_ASYNC
#define WAIT_ASYNC(N) asm volatile("s_wait_asynccnt %0" ::"n"(N) : "memory")
#endif

// One async DMA: each of 32 lanes copies 16B global -> LDS (512B/instr).
__device__ __forceinline__ void async_b128(unsigned lds, const void* g) {
  asm volatile("global_load_async_to_lds_b128 %0, %1, off" ::"v"(lds), "v"(g)
               : "memory");
}

__device__ __forceinline__ unsigned lds_off(const void* p) {
  return (unsigned)(uintptr_t)p;  // flat LDS addr: low 32 bits = LDS offset
}

__device__ __forceinline__ void cvt8(v16bf& a, int base, float4 u, float4 w) {
  a[base + 0] = (__bf16)u.x; a[base + 1] = (__bf16)u.y;
  a[base + 2] = (__bf16)u.z; a[base + 3] = (__bf16)u.w;
  a[base + 4] = (__bf16)w.x; a[base + 5] = (__bf16)w.y;
  a[base + 6] = (__bf16)w.z; a[base + 7] = (__bf16)w.w;
}

// ---------------------------------------------------------------------------
// Prep A: K (B,S,D) f32 -> Kt (B,D,S) bf16 (LDS-tiled 64x64 transpose)
// ---------------------------------------------------------------------------
__global__ __launch_bounds__(256) void prep_k_kernel(
    const float* __restrict__ k, __bf16* __restrict__ kt) {
  constexpr int S = 2048, D = 256;
  const int b  = blockIdx.x;
  const int s0 = blockIdx.y * 64;
  const int d0 = blockIdx.z * 64;
  __shared__ float tile[64][65];
  const int t = threadIdx.x;

  const int dx  = (t & 15) * 4;
  const int syb = t >> 4;
#pragma unroll
  for (int p = 0; p < 4; ++p) {
    const int sy = syb + p * 16;
    float4 x = *(const float4*)(k + (size_t)(b * S + s0 + sy) * D + d0 + dx);
    tile[sy][dx + 0] = x.x; tile[sy][dx + 1] = x.y;
    tile[sy][dx + 2] = x.z; tile[sy][dx + 3] = x.w;
  }
  __syncthreads();

  const int dy = t & 63;
  const int sg = (t >> 6) * 16;
  v16bf o;
#pragma unroll
  for (int i = 0; i < 16; ++i) o[i] = (__bf16)tile[sg + i][dy];
  *(v16bf*)(kt + (size_t)(b * D + d0 + dy) * S + s0 + sg) = o;
}

// ---------------------------------------------------------------------------
// Prep B: V f32 -> Vb bf16 (vectorized streaming convert)
// ---------------------------------------------------------------------------
__global__ __launch_bounds__(256) void prep_v_kernel(
    const float* __restrict__ v, __bf16* __restrict__ vb) {
  const size_t i = ((size_t)blockIdx.x * 256 + threadIdx.x) * 8;
  float4 x0 = *(const float4*)(v + i);
  float4 x1 = *(const float4*)(v + i + 4);
  v16bf pk;
  cvt8(pk, 0, x0, x1);
  __bf16* o = vb + i;
#pragma unroll
  for (int e = 0; e < 8; ++e) o[e] = pk[e];
}

// ---------------------------------------------------------------------------
// Prep C: weights -> Wt[st][k(0..511)][n(0..255)] bf16, transposed.
// ---------------------------------------------------------------------------
__global__ __launch_bounds__(256) void prep_w_kernel(
    const float* __restrict__ Wi0, const float* __restrict__ Wi1,
    const float* __restrict__ Wo0, const float* __restrict__ Wo1,
    __bf16* __restrict__ wt) {
  constexpr int D = 256;
  const int m    = blockIdx.x;
  const int st   = m >> 1;
  const int half = m & 1;
  const float* src = half ? (st ? Wo1 : Wo0) : (st ? Wi1 : Wi0);
  const int n0 = blockIdx.y * 64;
  const int k0 = blockIdx.z * 64;
  __shared__ float tile[64][65];
  const int t = threadIdx.x;

  const int kx  = (t & 15) * 4;
  const int nyb = t >> 4;
#pragma unroll
  for (int p = 0; p < 4; ++p) {
    const int ny = nyb + p * 16;
    float4 x = *(const float4*)(src + (size_t)(n0 + ny) * D + k0 + kx);
    tile[ny][kx + 0] = x.x; tile[ny][kx + 1] = x.y;
    tile[ny][kx + 2] = x.z; tile[ny][kx + 3] = x.w;
  }
  __syncthreads();

  const int ky = t & 63;
  const int ng = (t >> 6) * 16;
  v16bf o;
#pragma unroll
  for (int i = 0; i < 16; ++i) o[i] = (__bf16)tile[ng + i][ky];
  *(v16bf*)(wt + (size_t)(st * 512 + half * 256 + k0 + ky) * 256 + n0 + ng) = o;
}

// ---------------------------------------------------------------------------
// Kernel 1: bidirectional flash attention with async global->LDS pipeline.
// Block = 64 threads (2 waves), 16 query rows. Wave w owns cols [w*128,+128).
// K tile double-buffered in LDS (async prefetch of next 32 keys overlaps the
// current tile's WMMAs); V tile pipelined within the iteration.
// ---------------------------------------------------------------------------
__global__ __launch_bounds__(64) void attn_bidir_kernel(
    const float* __restrict__ q, const __bf16* __restrict__ kt,
    const __bf16* __restrict__ vb, float* __restrict__ ws_fw,
    float* __restrict__ ws_bw) {
  constexpr int S = 2048, D = 256;
  const int lane = threadIdx.x & 31;
  const int wave = threadIdx.x >> 5;
  const int b    = blockIdx.x >> 7;
  const int i0   = (blockIdx.x & 127) << 4;

  // K rows padded 32->40 bf16 (80B, 16B aligned) to spread LDS banks.
  __shared__ __align__(16) __bf16 kbuf[2][256][40];
  // V rows padded 256->264 bf16 (528B, 16B aligned).
  __shared__ __align__(16) __bf16 vbuf[32][264];
  __shared__ float  sc[16][33];
  __shared__ __bf16 pfw[16][32];
  __shared__ __bf16 pbw[16][32];
  __shared__ float  smf[16], ssf[16], smb[16], ssb[16];
  __shared__ float  ffw[16], fbw[16];

  const float scale = 0.0625f;  // 1/sqrt(256)
  const int mrow = lane & 15;
  const int koff = (lane >> 4) * 8;

  const __bf16* ktb = kt + (size_t)b * D * S;
  const __bf16* vbb = vb + (size_t)b * S * D;
  const unsigned kb0 = lds_off(&kbuf[0][0][0]);
  const unsigned kb1 = lds_off(&kbuf[1][0][0]);
  const unsigned vb0 = lds_off(&vbuf[0][0]);

  // async staging geometry (per wave: 16 instr per K tile, 16 per V tile)
  const int kdrow = wave * 128 + (lane >> 2);  // + i*8   (d rows)
  const int kchnk = (lane & 3);                // 16B chunk within 32 keys
  const int vkey0 = wave * 16;                 // + i     (key rows)

  // ---- Q tile (16x256) once, f32 -> eight bf16 A-tiles ----
  const float* qrow = q + (size_t)(b * S + i0 + mrow) * D;
  v16bf aq[8];
#pragma unroll
  for (int j = 0; j < 8; ++j) {
    const float* p0 = qrow + 32 * j + koff;
    const float* p1 = qrow + 32 * j + 16 + koff;
    float4 x0 = *(const float4*)p0, x1 = *(const float4*)(p0 + 4);
    float4 y0 = *(const float4*)p1, y1 = *(const float4*)(p1 + 4);
    cvt8(aq[j], 0, x0, x1);
    cvt8(aq[j], 8, y0, y1);
  }

  if (threadIdx.x < 16) {
    smf[threadIdx.x] = -INFINITY; smb[threadIdx.x] = -INFINITY;
    ssf[threadIdx.x] = 0.f;       ssb[threadIdx.x] = 0.f;
    ffw[threadIdx.x] = 1.f;       fbw[threadIdx.x] = 1.f;
  }

  v8f ofw[8], obw[8];
  const v8f vzero = {};
#pragma unroll
  for (int t = 0; t < 8; ++t) { ofw[t] = vzero; obw[t] = vzero; }

  const int dbase = wave * 128;

  // ---- prologue: async prefetch K tile 0 into kbuf[0] ----
  {
    const __bf16* g = ktb + (size_t)kdrow * S + kchnk * 8;
    unsigned l = kb0 + kdrow * 80 + kchnk * 16;
#pragma unroll
    for (int i = 0; i < 16; ++i) {
      async_b128(l, g);
      g += 8 * S;
      l += 8 * 80;
    }
  }
  __syncthreads();  // covers smf/ssf init too

  for (int j0 = 0; j0 < S; j0 += 32) {
    const int  cur      = (j0 >> 5) & 1;
    const bool has_next = (j0 + 32) < S;
    const bool fw_act   = (j0 + 31 >= i0);
    const bool bw_act   = (j0 <= i0 + 15);

    // ---- issue async: V(t) into vbuf, K(t+1) into other kbuf ----
    {
      const __bf16* g = vbb + (size_t)(j0 + vkey0) * D + lane * 8;
      unsigned l = vb0 + vkey0 * 528 + lane * 16;
#pragma unroll
      for (int i = 0; i < 16; ++i) {
        async_b128(l, g);
        g += D;
        l += 528;
      }
    }
    if (has_next) {
      const __bf16* g = ktb + (size_t)kdrow * S + (j0 + 32) + kchnk * 8;
      unsigned l = (cur ? kb0 : kb1) + kdrow * 80 + kchnk * 16;
#pragma unroll
      for (int i = 0; i < 16; ++i) {
        async_b128(l, g);
        g += 8 * S;
        l += 8 * 80;
      }
      WAIT_ASYNC(32);  // K(t) landed (in-order completion)
    } else {
      WAIT_ASYNC(16);  // K(t) landed
    }
    __syncthreads();   // all waves' K(t) visible

    // ---- scores for keys [j0 + wave*16, +16) from LDS ----
    v8f c = vzero;
    const __bf16* klds = &kbuf[cur][0][wave * 16];
#pragma unroll
    for (int j = 0; j < 8; ++j) {
      const int d = 32 * j;  // B layout: lane%16 = key col, elements = d (table order)
      v16bf bk;
#pragma unroll
      for (int s = 0; s < 16; ++s)
        bk[s] = klds[(d + (s >> 3) * 16 + (s & 7) + koff) * 40 + (lane & 15)];
      c = WMMA_BF16(aq[j], bk, c);
    }
#pragma unroll
    for (int r = 0; r < 8; ++r)
      sc[(lane >> 4) * 8 + r][wave * 16 + (lane & 15)] = c[r] * scale;
    __syncthreads();

    // ---- dual online softmax, single writer ----
    if (wave == 0 && lane < 16) {
      const int r = lane;
      const int row_i = i0 + r;
      float mf = smf[r], mb = smb[r];
      const float mfo = mf, mbo = mb;
#pragma unroll
      for (int kk = 0; kk < 32; ++kk) {
        const float s = sc[r][kk];
        const int jg = j0 + kk;
        if (fw_act && jg >= row_i) mf = fmaxf(mf, s);
        if (bw_act && jg <= row_i) mb = fmaxf(mb, s);
      }
      const float ff = (mf > mfo) ? __expf(mfo - mf) : 1.f;
      const float fb = (mb > mbo) ? __expf(mbo - mb) : 1.f;
      float sf = ssf[r] * ff, sb = ssb[r] * fb;
#pragma unroll
      for (int kk = 0; kk < 32; ++kk) {
        const float s = sc[r][kk];
        const int jg = j0 + kk;
        const float pf = (fw_act && jg >= row_i) ? __expf(s - mf) : 0.f;
        const float pb = (bw_act && jg <= row_i) ? __expf(s - mb) : 0.f;
        sf += pf; sb += pb;
        pfw[r][kk] = (__bf16)pf;
        pbw[r][kk] = (__bf16)pb;
      }
      smf[r] = mf; ssf[r] = sf; smb[r] = mb; ssb[r] = sb;
      ffw[r] = ff; fbw[r] = fb;
    }
    if (has_next) { WAIT_ASYNC(16); } else { WAIT_ASYNC(0); }  // V(t) landed
    __syncthreads();  // softmax results + all waves' V(t) visible

    // ---- P back as A-tiles; rescale + PV from LDS ----
    v16bf apf, apb;
#pragma unroll
    for (int h = 0; h < 16; ++h) {
      const int kk = (h >> 3) * 16 + (h & 7) + koff;
      apf[h] = pfw[mrow][kk];
      apb[h] = pbw[mrow][kk];
    }
    float frw[8], brw[8];
#pragma unroll
    for (int r = 0; r < 8; ++r) {
      const int m = (lane >> 4) * 8 + r;
      frw[r] = ffw[m]; brw[r] = fbw[m];
    }
    const __bf16* vlds = &vbuf[0][dbase + (lane & 15)];
#pragma unroll
    for (int nt = 0; nt < 8; ++nt) {
      v16bf bv;  // lane%16 = d-col, elements = keys (table order)
#pragma unroll
      for (int s = 0; s < 16; ++s)
        bv[s] = vlds[((s >> 3) * 16 + (s & 7) + koff) * 264 + nt * 16];
      if (fw_act) {
#pragma unroll
        for (int r = 0; r < 8; ++r) ofw[nt][r] *= frw[r];
        ofw[nt] = WMMA_BF16(apf, bv, ofw[nt]);
      }
      if (bw_act) {
#pragma unroll
        for (int r = 0; r < 8; ++r) obw[nt][r] *= brw[r];
        obw[nt] = WMMA_BF16(apb, bv, obw[nt]);
      }
    }
    __syncthreads();  // protect vbuf/sc/pfw for next iteration
  }

  // ---- normalize, store fw_v / bw_v ----
  float invf[8], invb[8];
#pragma unroll
  for (int r = 0; r < 8; ++r) {
    const int m = (lane >> 4) * 8 + r;
    invf[r] = 1.f / ssf[m];
    invb[r] = 1.f / ssb[m];
  }
#pragma unroll
  for (int nt = 0; nt < 8; ++nt) {
#pragma unroll
    for (int r = 0; r < 8; ++r) {
      const int m   = (lane >> 4) * 8 + r;
      const int col = dbase + nt * 16 + (lane & 15);
      const size_t idx = (size_t)(b * S + i0 + m) * D + col;
      ws_fw[idx] = ofw[nt][r] * invf[r];
      ws_bw[idx] = obw[nt][r] * invb[r];
    }
  }
}

// ---------------------------------------------------------------------------
// Kernel 2: gating. logits = [v | att] @ Wt_st + bo; out = g*att + (1-g)*v.
// ---------------------------------------------------------------------------
__global__ __launch_bounds__(32) void gate_fuse_kernel(
    const float* __restrict__ v, const float* __restrict__ att_fw,
    const float* __restrict__ att_bw, const __bf16* __restrict__ wt,
    const float* __restrict__ bo0, const float* __restrict__ bo1,
    float* __restrict__ out) {
  constexpr int D = 256;
  const int lane = threadIdx.x & 31;
  const int st   = blockIdx.y;
  const float* att = st ? att_bw : att_fw;
  const float* bo  = st ? bo1 : bo0;
  const __bf16* w  = wt + (size_t)st * 512 * 256;
  const int row0 = blockIdx.x * 16;
  const int mrow = lane & 15;
  const int koff = (lane >> 4) * 8;

  v8f acc[16];
  const v8f vzero = {};
#pragma unroll
  for (int t = 0; t < 16; ++t) acc[t] = vzero;

  for (int kc = 0; kc < 16; ++kc) {
    const float* xsrc = (kc < 8) ? v : att;
    const int kb = (kc & 7) * 32;
    const float* xrow = xsrc + (size_t)(row0 + mrow) * D + kb;
    float4 x0 = *(const float4*)(xrow + koff);
    float4 x1 = *(const float4*)(xrow + koff + 4);
    float4 y0 = *(const float4*)(xrow + 16 + koff);
    float4 y1 = *(const float4*)(xrow + 16 + koff + 4);
    v16bf a;
    cvt8(a, 0, x0, x1);
    cvt8(a, 8, y0, y1);
    const __bf16* wrow = w + (size_t)(kc * 32) * 256 + (lane & 15);
#pragma unroll
    for (int nt = 0; nt < 16; ++nt) {
      v16bf bw_;
#pragma unroll
      for (int s = 0; s < 16; ++s)
        bw_[s] = wrow[((s >> 3) * 16 + (s & 7) + koff) * 256 + nt * 16];
      acc[nt] = WMMA_BF16(a, bw_, acc[nt]);
    }
  }

#pragma unroll
  for (int nt = 0; nt < 16; ++nt) {
#pragma unroll
    for (int r = 0; r < 8; ++r) {
      const int m   = (lane >> 4) * 8 + r;
      const int col = nt * 16 + (lane & 15);
      const size_t t = (size_t)(row0 + m);
      const float logit = acc[nt][r] + bo[col];
      const float g  = 1.f / (1.f + __expf(-logit));
      const float vv = v[t * D + col];
      const float av = att[t * D + col];
      out[t * 2 * D + (size_t)st * D + col] = g * av + (1.f - g) * vv;
    }
  }
}

extern "C" void kernel_launch(void* const* d_in, const int* in_sizes, int n_in,
                              void* d_out, int out_size, void* d_ws,
                              size_t ws_size, hipStream_t stream) {
  constexpr int B = 8, S = 2048, D = 256;
  const float* q   = (const float*)d_in[0];
  const float* k   = (const float*)d_in[1];
  const float* v   = (const float*)d_in[2];
  const float* Wi0 = (const float*)d_in[3];
  const float* Wi1 = (const float*)d_in[4];
  const float* Wo0 = (const float*)d_in[5];
  const float* Wo1 = (const float*)d_in[6];
  const float* bo0 = (const float*)d_in[7];
  const float* bo1 = (const float*)d_in[8];
  float* out = (float*)d_out;

  char* p = (char*)d_ws;
  float* ws_fw = (float*)p;  p += (size_t)B * S * D * 4;
  float* ws_bw = (float*)p;  p += (size_t)B * S * D * 4;
  __bf16* Kt   = (__bf16*)p; p += (size_t)B * S * D * 2;   // (B, D, S)
  __bf16* Vb   = (__bf16*)p; p += (size_t)B * S * D * 2;   // (B, S, D)
  __bf16* Wt   = (__bf16*)p;                               // (2, 512, 256)

  prep_k_kernel<<<dim3(B, S / 64, D / 64), dim3(256), 0, stream>>>(k, Kt);
  prep_v_kernel<<<dim3((int)((size_t)B * S * D / 8 / 256)), dim3(256), 0,
                  stream>>>(v, Vb);
  prep_w_kernel<<<dim3(4, D / 64, D / 64), dim3(256), 0, stream>>>(
      Wi0, Wi1, Wo0, Wo1, Wt);

  attn_bidir_kernel<<<dim3(B * S / 16), dim3(64), 0, stream>>>(q, Kt, Vb,
                                                               ws_fw, ws_bw);

  gate_fuse_kernel<<<dim3(B * S / 16, 2), dim3(32), 0, stream>>>(
      v, ws_fw, ws_bw, Wt, bo0, bo1, out);
}
